// DeformLSTM_4999341933075
// MI455X (gfx1250) — matrix operand order repaired and checked by reading, not compiled, not run on hardware
//
#include <hip/hip_runtime.h>
#include <hip/hip_bf16.h>

// ---------------------------------------------------------------------------
// DeformLSTM on MI455X (gfx1250). All convs/FCs = implicit GEMM through
// v_wmma_f32_16x16x32_bf16 (wave32). All GEMM operands are padded by
// construction (K to 32, weights packed to bf16 [Mpad64][Kpad32], N always a
// multiple of 64) so the hot loop has ZERO bounds checks: one b128 global load
// per thread for A, one v8bf for B, vectorized ds_load_b128 fragment reads,
// two v_wmma per wave per K-chunk.
//
// d_ws layout (~158MB):
//   [0,24MB)      act ping (f32, max 18.9MB)
//   [24,48MB)     act pong
//   [48,136MB)    bf16 col buffer [Kpad][N] (max 576*73728*2 = 84.9MB)
//   [136,142MB)   offset maps (f32, max 5.3MB)
//   [142,144MB)   LSTM out bf16 [2560][128]
//   [144,146MB)   fc1 out bf16 [416][128]
//   [146,150MB)   packed offset-conv weights bf16 [64][Kpad]
//   [150,154MB)   packed deform-conv weights bf16 [co][Kpad]
//   [154,157MB)   packed fc1 weights bf16 [448][2560]
//   [157,158MB)   packed fc2 weights bf16 [64][416]
// ---------------------------------------------------------------------------

typedef __bf16 bf16_t;
typedef __attribute__((ext_vector_type(16))) __bf16 v16bf;
typedef __attribute__((ext_vector_type(8)))  __bf16 v8bf;
typedef __attribute__((ext_vector_type(8)))  float  v8f;

#define GT_M 64
#define GT_N 64
#define GT_K 32
#define LDSP (GT_K + 8)   // LDS row stride (bf16): 40*2=80B -> conflict-free rows

// epilogue modes
#define EPI_BIAS      0
#define EPI_BN_RELU   1
#define EPI_BIAS_RELU 2
// output modes
#define OUT_KN   0   // Out[m*N + n]
#define OUT_NM   1   // Out[n*M + m]
#define OUT_CONV 2   // Out[((n/HWp)*M + m)*HWp + n%HWp]  (NCHW scatter)

__device__ __forceinline__ void epi_store(
    void* Out, float v, int m, int n, int M, int N,
    int epi, int outMode, int HWp, int outBf16,
    const float* v0, const float* v1, const float* v2, const float* v3)
{
    if (m >= M) return;                    // N is exact by construction
    if (epi == EPI_BIAS) {
        v += v0[m];
    } else if (epi == EPI_BN_RELU) {
        float s = v0[m] * rsqrtf(v3[m] + 1e-5f);
        v = fmaxf((v - v2[m]) * s + v1[m], 0.f);
    } else if (epi == EPI_BIAS_RELU) {
        v = fmaxf(v + v0[m], 0.f);
    }
    size_t o;
    if (outMode == OUT_KN)      o = (size_t)m * N + n;
    else if (outMode == OUT_NM) o = (size_t)n * M + m;
    else { int b = n / HWp, p = n % HWp; o = ((size_t)b * M + m) * HWp + p; }
    if (outBf16) ((bf16_t*)Out)[o] = (bf16_t)v;
    else         ((float*)Out)[o]  = v;
}

// ---------------------------------------------------------------------------
// Out = A(bf16, packed [>=gridY*64][Kpad]) * B(bf16, [Kpad][N], N%64==0)
// 256 threads = 8 waves, macro tile 64x64; wave (wm,wn) computes rows
// wm*16..+15, cols wn*32..+31 as two 16x16 WMMA tiles sharing the A fragment.
// ---------------------------------------------------------------------------
__global__ __launch_bounds__(256) void wmma_gemm_kernel(
    const bf16_t* __restrict__ A, const bf16_t* __restrict__ B, void* __restrict__ Out,
    int M, int N, int Kpad, int epi, int outMode, int HWp, int outBf16,
    const float* __restrict__ v0, const float* __restrict__ v1,
    const float* __restrict__ v2, const float* __restrict__ v3)
{
    __shared__ bf16_t As[GT_M][LDSP];   // [m][k]
    __shared__ bf16_t Bt[GT_N][LDSP];   // [n][k] (transposed on stage-in)

    const int tid   = threadIdx.x;
    const int wave  = tid >> 5, lane = tid & 31;
    const int waveM = wave >> 1;        // 0..3
    const int waveN = wave & 1;         // 0..1
    const int lhalf = lane >> 4;        // K-half selector (A/B), M-half (C/D)
    const int lrow  = lane & 15;

    const int tileM = blockIdx.y * GT_M;
    const int tileN = blockIdx.x * GT_N;

    v8f acc0 = {}, acc1 = {};

    const int a_r = tid >> 2, a_c = (tid & 3) * 8;   // 64x32, 8 bf16/thread
    const int b_k = tid >> 3, b_n = (tid & 7) * 8;   // 32x64, 8 bf16/thread

    for (int k0 = 0; k0 < Kpad; k0 += GT_K) {
        // stage A: one 16B load per thread, no guards (A is padded)
        *(v8bf*)&As[a_r][a_c] =
            *(const v8bf*)&A[(size_t)(tileM + a_r) * Kpad + k0 + a_c];
        // stage B transposed: one 16B load per thread, scatter into Bt[n][k]
        v8bf bv = *(const v8bf*)&B[(size_t)(k0 + b_k) * N + tileN + b_n];
        #pragma unroll
        for (int j = 0; j < 8; ++j) Bt[b_n + j][b_k] = bv[j];
        if (k0 + GT_K < Kpad)
            __builtin_prefetch((const void*)&B[(size_t)(k0 + GT_K + b_k) * N + tileN + b_n], 0, 1);
        __syncthreads();

        // A fragment (ISA 7.12.2, 16-bit A 16x32): lane<16 -> K 0..7 & 16..23,
        // lane>=16 -> K 8..15 & 24..31, row = M = lrow. Two ds_load_b128.
        const bf16_t* arow = &As[waveM * 16 + lrow][lhalf * 8];
        v8bf a_lo = *(const v8bf*)arow;
        v8bf a_hi = *(const v8bf*)(arow + 16);
        v16bf af = __builtin_shufflevector(a_lo, a_hi,
                       0,1,2,3,4,5,6,7,8,9,10,11,12,13,14,15);

        // B fragment (16-bit B 32x16): lane<16 -> K 0..7 & 16..23, lane>=16 -> K 8..15 & 24..31,
        // col N = lrow. Two ds_load_b128 per tile from Bt row.
        const bf16_t* brow0 = &Bt[waveN * 32 + lrow][lhalf * 8];
        v8bf b0l = *(const v8bf*)brow0;
        v8bf b0h = *(const v8bf*)(brow0 + 16);
        v16bf bf0 = __builtin_shufflevector(b0l, b0h,
                        0,1,2,3,4,5,6,7,8,9,10,11,12,13,14,15);
        const bf16_t* brow1 = &Bt[waveN * 32 + 16 + lrow][lhalf * 8];
        v8bf b1l = *(const v8bf*)brow1;
        v8bf b1h = *(const v8bf*)(brow1 + 16);
        v16bf bf1 = __builtin_shufflevector(b1l, b1h,
                        0,1,2,3,4,5,6,7,8,9,10,11,12,13,14,15);

        acc0 = __builtin_amdgcn_wmma_f32_16x16x32_bf16(
                   false, af, false, bf0, (short)0, acc0, false, false);
        acc1 = __builtin_amdgcn_wmma_f32_16x16x32_bf16(
                   false, af, false, bf1, (short)0, acc1, false, false);
        __syncthreads();
    }

    // C/D layout: lanes 0-15: VGPR r -> M=r, N=lane; lanes 16-31: M=8+r.
    const int mb = tileM + waveM * 16 + lhalf * 8;
    const int n0 = tileN + waveN * 32 + lrow;
    #pragma unroll
    for (int r = 0; r < 8; ++r) {
        epi_store(Out, acc0[r], mb + r, n0,      M, N, epi, outMode, HWp, outBf16, v0, v1, v2, v3);
        epi_store(Out, acc1[r], mb + r, n0 + 16, M, N, epi, outMode, HWp, outBf16, v0, v1, v2, v3);
    }
}

// ---------------------------------------------------------------------------
// Weight pack: f32 [M][K] -> bf16 [Mpad][Kpad], zero padded
// ---------------------------------------------------------------------------
__global__ void pack_weights_kernel(const float* __restrict__ A, bf16_t* __restrict__ Ap,
                                    int M, int K, int Mpad, int Kpad)
{
    long total = (long)Mpad * Kpad;
    for (long idx = (long)blockIdx.x * blockDim.x + threadIdx.x; idx < total;
         idx += (long)gridDim.x * blockDim.x) {
        int m = (int)(idx / Kpad), k = (int)(idx % Kpad);
        float v = (m < M && k < K) ? A[(size_t)m * K + k] : 0.f;
        Ap[idx] = (bf16_t)v;
    }
}

__global__ void fill_zero_bf16_kernel(bf16_t* __restrict__ p, long n)
{
    for (long i = (long)blockIdx.x * blockDim.x + threadIdx.x; i < n;
         i += (long)gridDim.x * blockDim.x)
        p[i] = (bf16_t)0.f;
}

// ---------------------------------------------------------------------------
// Plain zero-pad 3x3 im2col: col[(ci*9+kk)*N + n] (bf16), N = B*H*W
// ---------------------------------------------------------------------------
__global__ void im2col_kernel(const float* __restrict__ x, bf16_t* __restrict__ col,
                              int Bn, int Ci, int H, int W)
{
    const int N = Bn * H * W;
    const long total = (long)Ci * 9 * N;
    for (long idx = (long)blockIdx.x * blockDim.x + threadIdx.x; idx < total;
         idx += (long)gridDim.x * blockDim.x) {
        int k  = (int)(idx / N);
        int n  = (int)(idx % N);
        int ci = k / 9, kk = k % 9, kh = kk / 3, kw = kk % 3;
        int b  = n / (H * W), p = n % (H * W), y = p / W, xx = p % W;
        int iy = y + kh - 1, ix = xx + kw - 1;
        float v = (iy >= 0 && iy < H && ix >= 0 && ix < W)
                      ? x[(((size_t)b * Ci + ci) * H + iy) * W + ix] : 0.f;
        col[idx] = (bf16_t)v;
    }
}

// ---------------------------------------------------------------------------
// Deformable bilinear gather -> im2col. idx = kk*N + n so adjacent lanes write
// adjacent col elements (coalesced) and read adjacent offsets. Loops channels.
// ---------------------------------------------------------------------------
__global__ void deform_im2col_kernel(const float* __restrict__ x, const float* __restrict__ off,
                                     bf16_t* __restrict__ col, int Bn, int Ci, int H, int W)
{
    const int N = Bn * H * W;
    long idx = (long)blockIdx.x * blockDim.x + threadIdx.x;
    if (idx >= (long)9 * N) return;
    int kk = (int)(idx / N);
    int n  = (int)(idx % N);
    int HW = H * W;
    int b = n / HW, p = n % HW, y = p / W, xx = p % W;

    float dX = off[((size_t)b * 18 + 2 * kk) * HW + p];      // even ch -> row offset
    float dY = off[((size_t)b * 18 + 2 * kk + 1) * HW + p];  // odd ch  -> col offset
    int ki = kk / 3, kj = kk % 3;
    float px = dX + (float)(ki - 1) + (float)(y + 1);
    float py = dY + (float)(kj - 1) + (float)(xx + 1);
    float fx = floorf(px), fy = floorf(py);
    int Hp = H + 2, Wp = W + 2;

    int x0 = (int)fminf(fmaxf(fx,       0.f), (float)(Hp - 1));
    int y0 = (int)fminf(fmaxf(fy,       0.f), (float)(Wp - 1));
    int x1 = (int)fminf(fmaxf(fx + 1.f, 0.f), (float)(Hp - 1));
    int y1 = (int)fminf(fmaxf(fy + 1.f, 0.f), (float)(Wp - 1));

    float pxs = (px < 1.f || px > (float)(Hp - 2)) ? fx : px;
    float pys = (py < 1.f || py > (float)(Wp - 2)) ? fy : py;
    pxs = fminf(fmaxf(pxs, 0.f), (float)(Hp - 1));
    pys = fminf(fmaxf(pys, 0.f), (float)(Wp - 1));

    float gxl = 1.f + ((float)x0 - pxs);
    float gxr = 1.f - ((float)x1 - pxs);
    float gyl = 1.f + ((float)y0 - pys);
    float gyr = 1.f - ((float)y1 - pys);
    float w00 = gxl * gyl, w11 = gxr * gyr, w01 = gxl * gyr, w10 = gxr * gyl;

    int ux0 = x0 - 1, ux1 = x1 - 1, uy0 = y0 - 1, uy1 = y1 - 1;
    bool v00 = (ux0 >= 0 && ux0 < H && uy0 >= 0 && uy0 < W);
    bool v11 = (ux1 >= 0 && ux1 < H && uy1 >= 0 && uy1 < W);
    bool v01 = (ux0 >= 0 && ux0 < H && uy1 >= 0 && uy1 < W);
    bool v10 = (ux1 >= 0 && ux1 < H && uy0 >= 0 && uy0 < W);

    for (int ci = 0; ci < Ci; ++ci) {
        const float* xb = x + ((size_t)b * Ci + ci) * HW;
        float s = 0.f;
        if (v00) s += w00 * xb[ux0 * W + uy0];
        if (v11) s += w11 * xb[ux1 * W + uy1];
        if (v01) s += w01 * xb[ux0 * W + uy1];
        if (v10) s += w10 * xb[ux1 * W + uy0];
        col[((size_t)(ci * 9 + kk)) * N + n] = (bf16_t)s;
    }
}

// ---------------------------------------------------------------------------
// 2x2 max pool, stride 2
// ---------------------------------------------------------------------------
__global__ void maxpool_kernel(const float* __restrict__ in, float* __restrict__ out,
                               int BC, int H, int W)
{
    int Ho = H >> 1, Wo = W >> 1;
    int total = BC * Ho * Wo;
    int t = blockIdx.x * blockDim.x + threadIdx.x;
    if (t >= total) return;
    int p  = t % (Ho * Wo);
    int bc = t / (Ho * Wo);
    int yo = p / Wo, xo = p % Wo;
    const float* ib = in + (size_t)bc * H * W;
    float a = ib[(2 * yo) * W + 2 * xo];
    float b = ib[(2 * yo) * W + 2 * xo + 1];
    float c = ib[(2 * yo + 1) * W + 2 * xo];
    float d = ib[(2 * yo + 1) * W + 2 * xo + 1];
    out[t] = fmaxf(fmaxf(a, b), fmaxf(c, d));
}

// ---------------------------------------------------------------------------
// Fused 2-layer LSTM (hidden=10, seq=256, feat=9) + tanh; one batch element
// per thread, weights in LDS; writes bf16 [K=t*10+j][N=batch] for fc1.
// ---------------------------------------------------------------------------
__device__ __forceinline__ float sigf(float v) { return 1.f / (1.f + __expf(-v)); }

__global__ __launch_bounds__(128) void lstm2_kernel(
    const float* __restrict__ act,
    const float* __restrict__ wih0, const float* __restrict__ whh0,
    const float* __restrict__ wih1, const float* __restrict__ whh1,
    bf16_t* __restrict__ fcin)
{
    __shared__ float s_ih0[40 * 9];
    __shared__ float s_hh0[40 * 10];
    __shared__ float s_ih1[40 * 10];
    __shared__ float s_hh1[40 * 10];
    int tid = threadIdx.x;
    for (int i = tid; i < 360; i += 128) s_ih0[i] = wih0[i];
    for (int i = tid; i < 400; i += 128) {
        s_hh0[i] = whh0[i]; s_ih1[i] = wih1[i]; s_hh1[i] = whh1[i];
    }
    __syncthreads();

    int b = tid;
    float h0[10] = {0}, c0[10] = {0}, h1[10] = {0}, c1[10] = {0};
    for (int t = 0; t < 256; ++t) {
        float xf[9];
        #pragma unroll
        for (int f = 0; f < 9; ++f) xf[f] = act[((size_t)b * 256 + t) * 9 + f];

        float z[40];
        #pragma unroll 4
        for (int g = 0; g < 40; ++g) {
            float a = 0.f;
            #pragma unroll
            for (int f = 0; f < 9; ++f)  a += s_ih0[g * 9 + f] * xf[f];
            #pragma unroll
            for (int j = 0; j < 10; ++j) a += s_hh0[g * 10 + j] * h0[j];
            z[g] = a;
        }
        #pragma unroll
        for (int j = 0; j < 10; ++j) {
            float ig = sigf(z[j]),       fg = sigf(z[10 + j]);
            float gg = tanhf(z[20 + j]), og = sigf(z[30 + j]);
            c0[j] = fg * c0[j] + ig * gg;
            h0[j] = og * tanhf(c0[j]);
        }
        #pragma unroll 4
        for (int g = 0; g < 40; ++g) {
            float a = 0.f;
            #pragma unroll
            for (int j = 0; j < 10; ++j) a += s_ih1[g * 10 + j] * h0[j];
            #pragma unroll
            for (int j = 0; j < 10; ++j) a += s_hh1[g * 10 + j] * h1[j];
            z[g] = a;
        }
        #pragma unroll
        for (int j = 0; j < 10; ++j) {
            float ig = sigf(z[j]),       fg = sigf(z[10 + j]);
            float gg = tanhf(z[20 + j]), og = sigf(z[30 + j]);
            c1[j] = fg * c1[j] + ig * gg;
            h1[j] = og * tanhf(c1[j]);
            fcin[((size_t)t * 10 + j) * 128 + b] = (bf16_t)tanhf(h1[j]);
        }
    }
}

// ---------------------------------------------------------------------------
// Host orchestration
// ---------------------------------------------------------------------------
static inline int capgrid(long total) {
    long b = (total + 255) / 256;
    return (int)(b > 65535 ? 65535 : (b < 1 ? 1 : b));
}

extern "C" void kernel_launch(void* const* d_in, const int* in_sizes, int n_in,
                              void* d_out, int out_size, void* d_ws, size_t ws_size,
                              hipStream_t stream)
{
    (void)in_sizes; (void)n_in; (void)out_size; (void)ws_size;

    static const int  CH_ci[8] = {1, 64, 64, 128, 128, 128, 256, 256};
    static const int  CH_co[8] = {64, 64, 128, 128, 128, 256, 256, 256};
    static const bool POOL[8]  = {false, true, false, false, true, false, false, true};
    const int Bn = 128;
    const size_t MB = 1024 * 1024;

    char* ws = (char*)d_ws;
    float*  actA  = (float*)(ws);
    float*  actB  = (float*)(ws + 24 * MB);
    bf16_t* col   = (bf16_t*)(ws + 48 * MB);
    float*  offm  = (float*)(ws + 136 * MB);
    bf16_t* fcin  = (bf16_t*)(ws + 142 * MB);
    bf16_t* fc1o  = (bf16_t*)(ws + 144 * MB);   // [416][128]
    bf16_t* woP   = (bf16_t*)(ws + 146 * MB);   // packed offset-conv W [64][Kpad]
    bf16_t* wdP   = (bf16_t*)(ws + 150 * MB);   // packed deform-conv W [co][Kpad]
    bf16_t* fc1wP = (bf16_t*)(ws + 154 * MB);   // [448][2560]
    bf16_t* fc2wP = (bf16_t*)(ws + 157 * MB);   // [64][416]

    const float* src = (const float*)d_in[0];
    float* bufs[2] = {actA, actB};
    int cur = 0;
    int H = 24, W = 24;

    for (int i = 0; i < 8; ++i) {
        const int ci = CH_ci[i], co = CH_co[i];
        const int N = Bn * H * W, Kc = ci * 9, HWp = H * W;
        const int Kpad = (Kc + 31) & ~31;
        const float* wo    = (const float*)d_in[1 + i * 7 + 0];
        const float* bo    = (const float*)d_in[1 + i * 7 + 1];
        const float* wd    = (const float*)d_in[1 + i * 7 + 2];
        const float* gamma = (const float*)d_in[1 + i * 7 + 3];
        const float* beta  = (const float*)d_in[1 + i * 7 + 4];
        const float* mean  = (const float*)d_in[1 + i * 7 + 5];
        const float* var   = (const float*)d_in[1 + i * 7 + 6];

        // --- offset conv: pack W, im2col (+pad rows), GEMM(18 x Kc) + bias ---
        pack_weights_kernel<<<capgrid((long)64 * Kpad), 256, 0, stream>>>(
            wo, woP, 18, Kc, 64, Kpad);
        im2col_kernel<<<capgrid((long)Kc * N), 256, 0, stream>>>(src, col, Bn, ci, H, W);
        if (Kpad > Kc)
            fill_zero_bf16_kernel<<<capgrid((long)(Kpad - Kc) * N), 256, 0, stream>>>(
                col + (size_t)Kc * N, (long)(Kpad - Kc) * N);
        {
            dim3 g(N / GT_N, 1);
            wmma_gemm_kernel<<<g, 256, 0, stream>>>(
                woP, col, (void*)offm, 18, N, Kpad, EPI_BIAS, OUT_CONV, HWp, 0,
                bo, nullptr, nullptr, nullptr);
        }

        // --- deform conv: pack W, bilinear gather im2col (+pad), GEMM + BN+ReLU
        pack_weights_kernel<<<capgrid((long)co * Kpad), 256, 0, stream>>>(
            wd, wdP, co, Kc, co, Kpad);   // co is a multiple of 64
        deform_im2col_kernel<<<capgrid((long)9 * N), 256, 0, stream>>>(
            src, offm, col, Bn, ci, H, W);
        if (Kpad > Kc)
            fill_zero_bf16_kernel<<<capgrid((long)(Kpad - Kc) * N), 256, 0, stream>>>(
                col + (size_t)Kc * N, (long)(Kpad - Kc) * N);
        float* dst = bufs[cur];
        {
            dim3 g(N / GT_N, co / GT_M);
            wmma_gemm_kernel<<<g, 256, 0, stream>>>(
                wdP, col, (void*)dst, co, N, Kpad, EPI_BN_RELU, OUT_CONV, HWp, 0,
                gamma, beta, mean, var);
        }
        src = dst; cur ^= 1;

        if (POOL[i]) {
            float* pdst = bufs[cur];
            int tot = Bn * co * (H / 2) * (W / 2);
            maxpool_kernel<<<(tot + 255) / 256, 256, 0, stream>>>(src, pdst, Bn * co, H, W);
            src = pdst; cur ^= 1;
            H /= 2; W /= 2;
        }
    }

    // --- LSTM x2 + tanh -> bf16 [2560][128] ---
    lstm2_kernel<<<1, 128, 0, stream>>>(
        src,
        (const float*)d_in[57], (const float*)d_in[58],
        (const float*)d_in[59], (const float*)d_in[60], fcin);

    // --- fc1: (400 x 2560) * (2560 x 128) + bias + ReLU -> bf16 [416][128] ---
    pack_weights_kernel<<<capgrid((long)448 * 2560), 256, 0, stream>>>(
        (const float*)d_in[61], fc1wP, 400, 2560, 448, 2560);
    fill_zero_bf16_kernel<<<capgrid((long)16 * 128), 256, 0, stream>>>(
        fc1o + (size_t)400 * 128, (long)16 * 128);   // pad rows 400..415 for fc2
    {
        dim3 g(128 / GT_N, 448 / GT_M);
        wmma_gemm_kernel<<<g, 256, 0, stream>>>(
            fc1wP, fcin, (void*)fc1o, 400, 128, 2560, EPI_BIAS_RELU, OUT_KN, 0, 1,
            (const float*)d_in[62], nullptr, nullptr, nullptr);
    }
    // --- fc2: (10 x 400) * (400 x 128) + bias -> d_out (b,10) f32 ---
    pack_weights_kernel<<<capgrid((long)64 * 416), 256, 0, stream>>>(
        (const float*)d_in[63], fc2wP, 10, 400, 64, 416);
    {
        dim3 g(128 / GT_N, 1);
        wmma_gemm_kernel<<<g, 256, 0, stream>>>(
            fc2wP, fc1o, d_out, 10, 128, 416, EPI_BIAS, OUT_NM, 0, 0,
            (const float*)d_in[64], nullptr, nullptr, nullptr);
    }
}
